// LSTM_36318243455658
// MI455X (gfx1250) — matrix operand order repaired and checked by reading, not compiled, not run on hardware
//
#include <hip/hip_runtime.h>

#define H        50
#define G        200        // 4*H, gate width
#define T_LEN    2048
#define MTILE    32         // batch rows per workgroup (2 x 16-row M-tiles)
#define TPB      256        // 8 wave32
#define NTILES   13         // ceil(208/16) covers G=200
#define KSTEPS   2          // K padded 50 -> 64
#define XCHUNK   128
#define GSTRIDE  212        // gates LDS row stride (floats)
#define HSTRIDE  64         // h LDS row stride (f16 elements)
#define NELEM    (MTILE * H)   // 1600 cell elements
#define EITER    7             // ceil(1600/256)

typedef __attribute__((ext_vector_type(16))) _Float16 v16h;
typedef __attribute__((ext_vector_type(8)))  _Float16 v8h;
typedef __attribute__((ext_vector_type(8)))  float    v8f;

__device__ __forceinline__ float fast_tanh(float x) {
#if __has_builtin(__builtin_amdgcn_tanhf)
    return __builtin_amdgcn_tanhf(x);          // v_tanh_f32
#elif __has_builtin(__builtin_amdgcn_tanh_f32)
    return __builtin_amdgcn_tanh_f32(x);
#else
    return 1.0f - 2.0f / (__expf(2.0f * x) + 1.0f);
#endif
}
__device__ __forceinline__ float fast_sig(float x) {
    return fmaf(fast_tanh(0.5f * x), 0.5f, 0.5f);   // 1 trans op per sigmoid
}

__global__ __launch_bounds__(TPB)
void lstm_wmma_kernel(const float* __restrict__ x,
                      const float* __restrict__ w_ih,
                      const float* __restrict__ w_hh,
                      const float* __restrict__ b_ih,
                      const float* __restrict__ b_hh,
                      const float* __restrict__ w_fc,
                      const float* __restrict__ b_fc,
                      float* __restrict__ out)
{
    __shared__ __align__(16) float    gates[MTILE * GSTRIDE];   // f32 gates [32 x 208+pad]
    __shared__ __align__(16) _Float16 hbuf[MTILE * HSTRIDE];    // h in f16, K-padded to 64
    __shared__ __align__(16) float    h32[MTILE * H];           // f32 h of last step (fc)
    __shared__ __align__(16) float    xbuf[MTILE * XCHUNK];     // staged x chunk

    const int tid   = threadIdx.x;
    const int lane  = tid & 31;
    const int wave  = tid >> 5;
    const int mhalf = wave >> 2;        // which 16-row M-tile this wave computes
    const int nbase = wave & 3;         // N-slot group
    const int b0    = blockIdx.x * MTILE;

    // zero h (covers K padding 50..63 permanently)
    for (int i = tid; i < MTILE * HSTRIDE; i += TPB) hbuf[i] = (_Float16)0.f;

    // ---------------- persistent B fragments: B[k][n] = W_hh[n][k] in f16 ----------
    // 16-bit B layout (32x16): N = lane&15 ; K runs [kb,+8) and [kb+16,+8), kb = lane<16?0:8
    const int ncol  = lane & 15;
    const int khalf = (lane < 16) ? 0 : 16;

    v16h Bfrag[4][KSTEPS];
    float wsum[4], bsum[4];
    #pragma unroll
    for (int i = 0; i < 4; ++i) {
        const int tn = nbase + 4 * i;             // N-tile index (wave-uniform)
        const int n  = tn * 16 + ncol;
        const bool nv = (tn < NTILES) && (n < G);
        wsum[i] = nv ? w_ih[n] : 0.f;
        bsum[i] = nv ? (b_ih[n] + b_hh[n]) : 0.f;
        #pragma unroll
        for (int s = 0; s < KSTEPS; ++s) {
            v16h bf;
            #pragma unroll
            for (int e = 0; e < 16; ++e) {
                const int k = s * 32 + (khalf >> 1) + (e < 8 ? e : e + 8);
                float v = (nv && k < H) ? w_hh[n * H + k] : 0.f;
                bf[e] = (_Float16)v;
            }
            Bfrag[i][s] = bf;
        }
    }

    // ---------------- hoisted per-thread elementwise mapping (loop-invariant) ------
    float creg[EITER];
    int   goff[EITER];      // gates offset of i-gate:  m*GSTRIDE + j
    int   hoff[EITER];      // hbuf offset:             m*HSTRIDE + j
    int   h32off[EITER];    // h32 offset:              m*H + j
    int   xoff[EITER];      // xbuf row offset:         m*XCHUNK
    bool  eval[EITER];
    #pragma unroll
    for (int kk = 0; kk < EITER; ++kk) {
        const int idx = tid + kk * TPB;
        const int m = idx / H;
        const int j = idx - m * H;
        creg[kk]   = 0.f;
        eval[kk]   = (idx < NELEM);
        goff[kk]   = m * GSTRIDE + j;
        hoff[kk]   = m * HSTRIDE + j;
        h32off[kk] = m * H + j;
        xoff[kk]   = m * XCHUNK;
    }

    // 16-bit A layout (16x32): M = lane&15 ; K runs [kb,+8) and [kb+16,+8), kb = lane<16?0:8
    const int arow   = mhalf * 16 + (lane & 15);
    const int akbase = (lane < 16) ? 0 : 8;
    const int mrowhi = ((lane < 16) ? 0 : 8) + mhalf * 16;   // C-layout row base

    __syncthreads();

    for (int t = 0; t < T_LEN; ++t) {
        // ------------- stage x chunk every 128 steps (coalesced float4) -----------
        if ((t & (XCHUNK - 1)) == 0) {
            const int m   = tid >> 3;          // 0..31
            const int seg = tid & 7;           // 8 segments of 16 floats
            const float4* src = (const float4*)(x + (size_t)(b0 + m) * T_LEN + t + seg * 16);
            float4*       dst = (float4*)(xbuf + m * XCHUNK + seg * 16);
            #pragma unroll
            for (int q = 0; q < 4; ++q) dst[q] = src[q];
            __syncthreads();
        }
        const int tc = t & (XCHUNK - 1);

        // ------------- phase A: gates = x*w_ih + b + h @ W_hh^T via WMMA ----------
        v16h afrag[KSTEPS];
        #pragma unroll
        for (int s = 0; s < KSTEPS; ++s) {
            const _Float16* p = hbuf + arow * HSTRIDE + s * 32 + akbase;
            union { v16h v; v8h h[2]; } au;
            au.h[0] = *(const v8h*)(p);        // ds_load_b128
            au.h[1] = *(const v8h*)(p + 16);   // ds_load_b128
            afrag[s] = au.v;
        }

        float xv[8];
        #pragma unroll
        for (int r = 0; r < 8; ++r)
            xv[r] = xbuf[(mrowhi + r) * XCHUNK + tc];

        #pragma unroll
        for (int i = 0; i < 4; ++i) {
            const int tn = nbase + 4 * i;
            if (tn < NTILES) {                  // wave-uniform guard: EXEC stays all-1
                v8f acc;
                #pragma unroll
                for (int r = 0; r < 8; ++r) acc[r] = fmaf(xv[r], wsum[i], bsum[i]);
                #pragma unroll
                for (int s = 0; s < KSTEPS; ++s)
                    acc = __builtin_amdgcn_wmma_f32_16x16x32_f16(
                              false, afrag[s], false, Bfrag[i][s],
                              (short)0, acc, false, false);
                const int nb = tn * 16 + ncol;
                #pragma unroll
                for (int r = 0; r < 8; ++r)
                    gates[(mrowhi + r) * GSTRIDE + nb] = acc[r];
            }
        }
        __syncthreads();

        // ------------- phase B: elementwise LSTM cell update ----------------------
        const bool last = (t == T_LEN - 1);
        #pragma unroll
        for (int kk = 0; kk < EITER; ++kk) {
            if (eval[kk]) {
                const float* gm = gates + goff[kk];
                const float ig = fast_sig(gm[0]);
                const float fg = fast_sig(gm[H]);
                const float gg = fast_tanh(gm[2 * H]);
                const float og = fast_sig(gm[3 * H]);
                const float c  = fg * creg[kk] + ig * gg;
                creg[kk] = c;
                const float h = og * fast_tanh(c);
                hbuf[hoff[kk]] = (_Float16)h;
                if (last) h32[h32off[kk]] = h;
            }
        }
        __syncthreads();
    }

    // ------------- fc head on last hidden state ----------------------------------
    if (tid < MTILE) {
        float acc = b_fc[0];
        #pragma unroll 5
        for (int j = 0; j < H; ++j) acc = fmaf(h32[tid * H + j], w_fc[j], acc);
        out[b0 + tid] = acc;
    }
}

extern "C" void kernel_launch(void* const* d_in, const int* in_sizes, int n_in,
                              void* d_out, int out_size, void* d_ws, size_t ws_size,
                              hipStream_t stream) {
    (void)n_in; (void)d_ws; (void)ws_size; (void)out_size;
    const float* x    = (const float*)d_in[0];
    const float* w_ih = (const float*)d_in[1];
    const float* w_hh = (const float*)d_in[2];
    const float* b_ih = (const float*)d_in[3];
    const float* b_hh = (const float*)d_in[4];
    const float* w_fc = (const float*)d_in[5];
    const float* b_fc = (const float*)d_in[6];
    float* out = (float*)d_out;

    const int batch = in_sizes[0] / T_LEN;     // INPUT_SIZE == 1
    const int grid  = batch / MTILE;           // 16384/32 = 512
    lstm_wmma_kernel<<<grid, TPB, 0, stream>>>(x, w_ih, w_hh, b_ih, b_hh, w_fc, b_fc, out);
}
